// MlpRBF_51908974739955
// MI455X (gfx1250) — matrix operand order repaired and checked by reading, not compiled, not run on hardware
//
#include <hip/hip_runtime.h>

typedef __attribute__((ext_vector_type(16))) _Float16     v16h;
typedef __attribute__((ext_vector_type(8)))  _Float16     v8h;
typedef __attribute__((ext_vector_type(2)))  _Float16     v2h;
typedef __attribute__((ext_vector_type(8)))  float        v8f;
typedef __attribute__((ext_vector_type(2)))  float        v2f;
typedef __attribute__((ext_vector_type(8)))  unsigned int v8u;
typedef __attribute__((ext_vector_type(4)))  unsigned int v4u;

#define B_ 16
#define T_ 1024
#define C_ 1024
#define H_ 16
#define WINDOW_ 0.25f
#define EPS_ 1e-5f
#define BT_PER_BLOCK 8

#if __has_builtin(__builtin_amdgcn_wmma_f16_16x16x32_f16)
#define HAS_WMMA_F16C 1
#else
#define HAS_WMMA_F16C 0
#endif

// Single-instruction ReLUs (avoid the v_max(x,x) canonicalize the compiler
// inserts for llvm.maxnum on WMMA results it cannot prove canonical).
__device__ __forceinline__ float relu_f32(float x) {
    float y;
    asm("v_max_num_f32 %0, %1, 0" : "=v"(y) : "v"(x));
    return y;
}
__device__ __forceinline__ unsigned relu_pk_f16(unsigned x) {
    unsigned y;
    asm("v_pk_max_num_f16 %0, %1, 0" : "=v"(y) : "v"(x));
    return y;
}

// 2048 blocks x 256 threads; each block processes 8 (b,t) pairs so the whole
// weight preamble is amortized 8x. Per (b,t): 8 waves x 4 iters of 32 rows =
// TWO 16-row tiles at once; every lane owns one row end-to-end.
// Dual-tile trick: a second A-operand (a1p/a2p) carries the same weights in
// the K=16..31 slots. One shared B register holds tile-A rows in lanes 0-15
// and tile-B rows in lanes 16-31; each WMMA's zero A-pad annihilates the half
// it doesn't own (0 * finite == 0). No EXEC masking anywhere in the hot loop.
// All four WMMAs per iteration use f16 C/D so results stay packed for the
// bpermute half-swaps and the packed-f16 layer-3 dot.
__global__ __launch_bounds__(256) void MlpRBF_51908974739955_kernel(
    const float* __restrict__ diff,
    const float* __restrict__ W0, const float* __restrict__ b0,
    const float* __restrict__ W1, const float* __restrict__ b1,
    const float* __restrict__ W2, const float* __restrict__ b2,
    const float* __restrict__ W3, const float* __restrict__ b3,
    float* __restrict__ out, float* __restrict__ dens_out)
{
    __shared__ float sred[8];
    __shared__ float sdens;

    const int tid  = threadIdx.x;
    const int lane = tid & 31;
    const int wave = tid >> 5;
    const int col  = lane & 15;
    const int hi   = lane >> 4;      // 0: lanes 0-15 (tile A), 1: lanes 16-31 (tile B)

    const v16h zh16 = {};

    // ---------------- preload weights (ONCE per block) ----------------
    // Layer-0 weights as packed f16 vectors (uniform). W0 is [2,16].
    v16h w0av = {}, w0bv = {}, b0vh = {};
#pragma unroll
    for (int j = 0; j < 16; ++j) {
        w0av[j] = (_Float16)W0[j];
        w0bv[j] = (_Float16)W0[16 + j];
        b0vh[j] = (_Float16)b0[j];
    }

    // A-operands: W^T in f16 A-layout. a1/a2 use K=0..15 (lanes 0-15, elems 0..15; lanes 16-31 pad 0);
    // a1p/a2p use K=16..31 (lanes 16-31, elems 0..15; lanes 0-15 pad 0).
    v16h a1 = {}, a2 = {}, a1p = {}, a2p = {};
#pragma unroll
    for (int e = 0; e < 16; ++e) {
        const _Float16 v1 = (_Float16)W1[e * H_ + col];
        const _Float16 v2 = (_Float16)W2[e * H_ + col];
        a1[e] = hi ? (_Float16)0 : v1;  a1p[e] = hi ? v1 : (_Float16)0;
        a2[e] = hi ? (_Float16)0 : v2;  a2p[e] = hi ? v2 : (_Float16)0;
    }

#if HAS_WMMA_F16C
    // Biases along M (output feature). f16 C/D: elem e -> M = e + 8*hi.
    v8h c1h = {}, c2h = {};
#pragma unroll
    for (int e = 0; e < 8; ++e) {
        c1h[e] = (_Float16)b1[hi * 8 + e];
        c2h[e] = (_Float16)b2[hi * 8 + e];
    }
    // Layer-3 weights, packed f16: this lane's half of W3.
    v8h w3p = {};
#pragma unroll
    for (int e = 0; e < 8; ++e) w3p[e] = (_Float16)W3[hi * 8 + e];
    const v4u w3q = __builtin_bit_cast(v4u, w3p);
#else
    v8f c1f = {}, c2f = {};
#pragma unroll
    for (int i = 0; i < 8; ++i) {
        c1f[i] = b1[hi * 8 + i];
        c2f[i] = b2[hi * 8 + i];
    }
    float w3h[8];
#pragma unroll
    for (int i = 0; i < 8; ++i) w3h[i] = W3[hi * 8 + i];
#endif
    const float b3v = b3[0];

    // ---------------- loop over this block's (b,t) pairs ----------------
#pragma unroll 1
    for (int btl = 0; btl < BT_PER_BLOCK; ++btl) {
        const int  bt      = blockIdx.x * BT_PER_BLOCK + btl;
        const long rowBase = (long)bt * C_;

        float wv[4];
        float acc = 0.0f;

        // ---- phase 1: 4 x 32 rows per wave ----
#pragma unroll
        for (int it = 0; it < 4; ++it) {
            const int base = it * 256 + wave * 32;   // this wave's 32 rows
            // lane n owns row base+n: coalesced 256B/wave, streamed (NT)
            const v2f xv = __builtin_nontemporal_load(
                ((const v2f*)diff) + (rowBase + base + lane));
            const float x0 = fabsf(xv.x);
            const float x1 = fabsf(xv.y);

            // Layer 0, full-wave: each lane emits its full 16-feature h0 row.
            // Lanes 0-15 -> K=0..15 (tile A), lanes 16-31 -> K=16..31 (tile B).
            const _Float16 x0h = (_Float16)x0;
            const _Float16 x1h = (_Float16)x1;
            v16h t0 = w0bv * x1h + b0vh;             // v_pk_fma_f16
            v16h h0 = w0av * x0h + t0;               // v_pk_fma_f16
            const v16h b1m = __builtin_elementwise_max(h0, zh16);

            float sA, sB;
#if HAS_WMMA_F16C
            // Layer 1 (transposed) for both tiles off the SAME B register.
            v8h dA = __builtin_amdgcn_wmma_f16_16x16x32_f16(
                false, a1,  false, b1m, (short)0, c1h, false, false);
            v8h dB = __builtin_amdgcn_wmma_f16_16x16x32_f16(
                false, a1p, false, b1m, (short)0, c1h, false, false);
            v4u ta = __builtin_bit_cast(v4u, dA);
            v4u tb = __builtin_bit_cast(v4u, dB);
            v4u qoA, qoB;
#pragma unroll
            for (int i = 0; i < 4; ++i) {
                qoA[i] = relu_pk_f16(ta[i]);
                qoB[i] = relu_pk_f16(tb[i]);
            }

            // Half-swaps -> B-layouts of H1^T.
            // Tile A rows in lanes 0-15 (K=0..15): [own | partner].
            // Tile B rows in lanes 16-31 (K=16..31): [partner | own].
            // Unused half-waves hold finite garbage killed by the A-pad.
            v8u qA, qB;
#pragma unroll
            for (int i = 0; i < 4; ++i) {
                qA[i]     = qoA[i];
                qA[4 + i] = (unsigned)__shfl_xor((int)qoA[i], 16, 32);
                qB[i]     = (unsigned)__shfl_xor((int)qoB[i], 16, 32);
                qB[4 + i] = qoB[i];
            }
            const v16h b2mA = __builtin_bit_cast(v16h, qA);
            const v16h b2mB = __builtin_bit_cast(v16h, qB);

            // Layer 2 (transposed, f16 C/D) for both tiles.
            v8h d2A = __builtin_amdgcn_wmma_f16_16x16x32_f16(
                false, a2,  false, b2mA, (short)0, c2h, false, false);
            v8h d2B = __builtin_amdgcn_wmma_f16_16x16x32_f16(
                false, a2p, false, b2mB, (short)0, c2h, false, false);

            // Layer 3: packed-f16 half-dots (4x v_pk_fma_f16 each).
            v4u ra = __builtin_bit_cast(v4u, d2A);
            v4u rb = __builtin_bit_cast(v4u, d2B);
            v2h accA = {}, accB = {};
#pragma unroll
            for (int i = 0; i < 4; ++i) {
                const v2h w2i = __builtin_bit_cast(v2h, w3q[i]);
                accA += __builtin_bit_cast(v2h, relu_pk_f16(ra[i])) * w2i;
                accB += __builtin_bit_cast(v2h, relu_pk_f16(rb[i])) * w2i;
            }
            sA = (float)accA[0] + (float)accA[1];
            sB = (float)accB[0] + (float)accB[1];
#else
            // Fallback: f32 accumulators throughout.
            v8f dA = __builtin_amdgcn_wmma_f32_16x16x32_f16(
                false, a1,  false, b1m, (short)0, c1f, false, false);
            v8f dB = __builtin_amdgcn_wmma_f32_16x16x32_f16(
                false, a1p, false, b1m, (short)0, c1f, false, false);
            v8h rA, rB;
#pragma unroll
            for (int i = 0; i < 8; ++i) {
                rA[i] = (_Float16)relu_f32(dA[i]);
                rB[i] = (_Float16)relu_f32(dB[i]);
            }
            v4u qoA = __builtin_bit_cast(v4u, rA);
            v4u qoB = __builtin_bit_cast(v4u, rB);
            v8u qA, qB;
#pragma unroll
            for (int i = 0; i < 4; ++i) {
                qA[i]     = qoA[i];
                qA[4 + i] = (unsigned)__shfl_xor((int)qoA[i], 16, 32);
                qB[i]     = (unsigned)__shfl_xor((int)qoB[i], 16, 32);
                qB[4 + i] = qoB[i];
            }
            const v16h b2mA = __builtin_bit_cast(v16h, qA);
            const v16h b2mB = __builtin_bit_cast(v16h, qB);
            v8f d2A = __builtin_amdgcn_wmma_f32_16x16x32_f16(
                false, a2,  false, b2mA, (short)0, c2f, false, false);
            v8f d2B = __builtin_amdgcn_wmma_f32_16x16x32_f16(
                false, a2p, false, b2mB, (short)0, c2f, false, false);
            sA = 0.0f; sB = 0.0f;
#pragma unroll
            for (int i = 0; i < 8; ++i) {
                sA = fmaf(relu_f32(d2A[i]), w3h[i], sA);
                sB = fmaf(relu_f32(d2B[i]), w3h[i], sB);
            }
#endif
            // lane n<16 needs sA(n)+sA(n+16); lane n>=16 needs sB(n)+sB(n-16).
            const float cross = __shfl_xor(hi ? sA : sB, 16, 32);
            const float h3 = (hi ? sB : sA) + cross + b3v;

            // weight = |mlp| * (|x0| < WINDOW); lane owns its row's mask.
            wv[it] = (x0 < WINDOW_) ? fabsf(h3) : 0.0f;
            acc += wv[it];
        }

        // ---- density: block reduction ----
#pragma unroll
        for (int m = 1; m <= 16; m <<= 1) acc += __shfl_xor(acc, m, 32);
        if (lane == 0) sred[wave] = acc;
        __syncthreads();
        if (tid == 0) {
            float s = 0.0f;
#pragma unroll
            for (int i = 0; i < 8; ++i) s += sred[i];
            sdens = s;
            dens_out[bt] = s;
        }
        __syncthreads();

        // ---- normalize + write (weights still in registers) ----
        const float inv = 1.0f / (sdens + EPS_);
#pragma unroll
        for (int it = 0; it < 4; ++it)
            __builtin_nontemporal_store(
                wv[it] * inv,
                out + (rowBase + it * 256 + wave * 32 + lane));
    }
}

extern "C" void kernel_launch(void* const* d_in, const int* in_sizes, int n_in,
                              void* d_out, int out_size, void* d_ws, size_t ws_size,
                              hipStream_t stream) {
    (void)in_sizes; (void)n_in; (void)out_size; (void)d_ws; (void)ws_size;
    const float* diff = (const float*)d_in[0];
    const float* W0   = (const float*)d_in[1];
    const float* b0   = (const float*)d_in[2];
    const float* W1   = (const float*)d_in[3];
    const float* b1   = (const float*)d_in[4];
    const float* W2   = (const float*)d_in[5];
    const float* b2   = (const float*)d_in[6];
    const float* W3   = (const float*)d_in[7];
    const float* b3   = (const float*)d_in[8];

    float* out  = (float*)d_out;
    float* dens = out + (size_t)B_ * T_ * C_;   // tuple output: [out | density]

    dim3 grid((B_ * T_) / BT_PER_BLOCK);        // 8 (b,t) pairs per block
    dim3 block(256);
    MlpRBF_51908974739955_kernel<<<grid, block, 0, stream>>>(
        diff, W0, b0, W1, b1, W2, b2, W3, b3, out, dens);
}
